// TMIL_87780541596192
// MI455X (gfx1250) — matrix operand order repaired and checked by reading, not compiled, not run on hardware
//
#include <hip/hip_runtime.h>
#include <math.h>

typedef __attribute__((ext_vector_type(16))) _Float16 v16h;
typedef __attribute__((ext_vector_type(8)))  _Float16 v8h;
typedef __attribute__((ext_vector_type(4)))  _Float16 v4h;
typedef __attribute__((ext_vector_type(8)))  float    v8f;
typedef __attribute__((ext_vector_type(4)))  float    v4f;
typedef __attribute__((ext_vector_type(2)))  float    v2f;

#define DMODEL 256
#define NTOK   40000
#define SIDE   200
#define NS     40001          // tokens incl cls
#define PADF   63             // (128 - NS%128) % 128
#define NPAD   40064          // NS + PADF = 313*128
#define LCH    313            // NPAD / 128
#define QSCALE 0.17677669529663687f  // 32^-0.5

// ---------------- wave32 reductions ----------------
__device__ inline float wsum(float v) {
#pragma unroll
  for (int o = 16; o > 0; o >>= 1) v += __shfl_xor(v, o, 32);
  return v;
}
__device__ inline float wmaxr(float v) {
#pragma unroll
  for (int o = 16; o > 0; o >>= 1) v = fmaxf(v, __shfl_xor(v, o, 32));
  return v;
}

// ---------------- async global->LDS copy (CDNA5 ASYNCcnt path) ----------------
// Per-lane 16B transfers; LDS address = low 32 bits of the generic pointer
// (flat LDS aperture maps addr[31:0] to the wave's LDS allocation).
__device__ inline void async_copy_f32(float* dst_lds, const float* src,
                                      int nfloats, int tid, int nthreads)
{
  for (int i = tid * 4; i < nfloats; i += nthreads * 4) {
    unsigned l = (unsigned)(uintptr_t)(dst_lds + i);
    const float* g = src + i;
    asm volatile("global_load_async_to_lds_b128 %0, %1, off"
                 :: "v"(l), "v"(g) : "memory");
  }
  asm volatile("s_wait_asynccnt 0" ::: "memory");
}

// ================= f16 WMMA GEMM =================
// C[M,N] = epilogue(A'[M,K] @ B[K,N]), A' row r = A[r - aoff] (zero outside [0,arows))
// Tiles staged into LDS in WMMA *fragment order*:
//  - A row stored in natural K order (lane half h reads K 16h..16h+15); a lane's 16
//    fragment halves are then 32 contiguous bytes -> 2x ds_load_b128.
//  - B stored column-major (Bt[col][k]); lane fragment = K 0..15 / 16..31 contiguous.
#define BM 128
#define BN 128
#define BKK 32
#define ASTRH 40   // halves per A row slot (80B, bank-spread)
#define BSTRH 40   // halves per B col slot

__global__ void __launch_bounds__(256)
k_gemm_f16(const float* __restrict__ A, int arows, int aoff, int lda,
           const float* __restrict__ B, int ldb,
           const float* __restrict__ bias,
           float* __restrict__ C, int ldc,
           float* __restrict__ Dacc, int doff, int ldd,
           int M, int N, int K, int relu, float qscale, int qcols)
{
  __shared__ __align__(16) _Float16 As2[BM * ASTRH];
  __shared__ __align__(16) _Float16 Bt[BN * BSTRH];
  const int m0 = blockIdx.x * BM, n0 = blockIdx.y * BN;
  const int tid = threadIdx.x, lane = tid & 31, wv = tid >> 5;

  v8f acc[8];
#pragma unroll
  for (int i = 0; i < 8; ++i)
#pragma unroll
    for (int e = 0; e < 8; ++e) acc[i][e] = 0.f;

  for (int k0 = 0; k0 < K; k0 += BKK) {
    // ---- stage A tile (128 rows x 32 K) as f16, natural K order ----
#pragma unroll
    for (int it = 0; it < 4; ++it) {
      int i = tid + it * 256;               // 0..1023
      int r = i >> 3, c4 = i & 7;
      int kk = c4 * 4;
      int gr = m0 + r, ar = gr - aoff;
      v4f vf; vf[0] = vf[1] = vf[2] = vf[3] = 0.f;
      if (gr < M && ar >= 0 && ar < arows && (k0 + kk) < K) {
        vf = *(const v4f*)(A + (long)ar * lda + k0 + kk);
        if (k0 + BKK + kk < K)
          __builtin_prefetch(A + (long)ar * lda + k0 + BKK + kk, 0, 1);
      }
      v4h hh;
#pragma unroll
      for (int e = 0; e < 4; ++e) hh[e] = (_Float16)vf[e];
      int g = kk >> 3;                       // K-group of 8
      int pg = (g == 1) ? 2 : ((g == 2) ? 1 : g);  // swap middle groups
      int pos = kk;
      *(v4h*)(As2 + r * ASTRH + pos) = hh;
    }
    // ---- stage B tile (32 K x 128 cols) transposed into Bt[col][k] ----
#pragma unroll
    for (int it = 0; it < 4; ++it) {
      int i = tid + it * 256;               // 0..1023
      int kk = i >> 5, c4 = (i & 31) * 4;
      v4f vf; vf[0] = vf[1] = vf[2] = vf[3] = 0.f;
      if ((k0 + kk) < K && (n0 + c4) < N)
        vf = *(const v4f*)(B + (long)(k0 + kk) * ldb + n0 + c4);
#pragma unroll
      for (int e = 0; e < 4; ++e) Bt[(c4 + e) * BSTRH + kk] = (_Float16)vf[e];
    }
    __syncthreads();

    // ---- fragments: contiguous 32B per lane -> 2x ds_load_b128 each ----
    const int mr = wv * 16 + (lane & 15);
    const int hbase = (lane < 16) ? 0 : 16;
    const _Float16* ap = As2 + mr * ASTRH + hbase;
    v8h alo = *(const v8h*)(ap);
    v8h ahi = *(const v8h*)(ap + 8);
    v16h af = __builtin_shufflevector(alo, ahi, 0,1,2,3,4,5,6,7,8,9,10,11,12,13,14,15);
    // preload all 8 B fragments first so the DS latency overlaps the WMMA chain
    v16h bfr[8];
#pragma unroll
    for (int nt = 0; nt < 8; ++nt) {
      int c = nt * 16 + (lane & 15);
      const _Float16* bp = Bt + c * BSTRH + hbase;
      v8h blo = *(const v8h*)(bp);
      v8h bhi = *(const v8h*)(bp + 8);
      bfr[nt] = __builtin_shufflevector(blo, bhi, 0,1,2,3,4,5,6,7,8,9,10,11,12,13,14,15);
    }
#pragma unroll
    for (int nt = 0; nt < 8; ++nt)
      acc[nt] = __builtin_amdgcn_wmma_f32_16x16x32_f16(false, af, false, bfr[nt],
                                                       (short)0, acc[nt], false, false);
    __syncthreads();
  }

#pragma unroll
  for (int nt = 0; nt < 8; ++nt) {
#pragma unroll
    for (int r = 0; r < 8; ++r) {
      int row = m0 + wv * 16 + r + (lane < 16 ? 0 : 8);
      int col = n0 + nt * 16 + (lane & 15);
      if (row < M && col < N) {
        float v = acc[nt][r];
        if (bias) v += bias[col];
        if (col < qcols) v *= qscale;
        if (relu) v = fmaxf(v, 0.f);
        if (Dacc) {
          int dr = row - doff;
          if (dr >= 0) Dacc[(long)dr * ldd + col] += v;
        } else {
          C[(long)row * ldc + col] = v;
        }
      }
    }
  }
}

// ================= fp32 WMMA 128x128 matmul (pinv Newton-Schulz) =================
// per head (blockIdx.x): C1 = s1*(A@B) + d1*I ; optional C2 = s2*(A@B) + d2*I
__global__ void __launch_bounds__(256)
k_mm128(const float* __restrict__ A, const float* __restrict__ B,
        float* __restrict__ C1, float s1, float d1,
        float* __restrict__ C2, float s2, float d2)
{
  const int hoff = blockIdx.x * (128 * 128);
  const float* Ah = A + hoff;
  const float* Bh = B + hoff;
  float* C1h = C1 + hoff;
  float* C2h = C2 ? (C2 + hoff) : nullptr;
  const int lane = threadIdx.x & 31, wv = threadIdx.x >> 5;

  v8f acc[8];
#pragma unroll
  for (int i = 0; i < 8; ++i)
#pragma unroll
    for (int e = 0; e < 8; ++e) acc[i][e] = 0.f;

  const int m = wv * 16 + (lane & 15);
  const int kb = (lane < 16) ? 0 : 2;
#if __has_builtin(__builtin_amdgcn_wmma_f32_16x16x4_f32)
  for (int k0 = 0; k0 < 128; k0 += 4) {
    v2f a = *(const v2f*)(Ah + m * 128 + k0 + kb);   // global_load_b64
#pragma unroll
    for (int nt = 0; nt < 8; ++nt) {
      v2f b;
      int c = nt * 16 + (lane & 15);
#pragma unroll
      for (int vv = 0; vv < 2; ++vv)
        b[vv] = Bh[(k0 + kb + vv) * 128 + c];
      acc[nt] = __builtin_amdgcn_wmma_f32_16x16x4_f32(false, a, false, b,
                                                      (short)0, acc[nt], false, false);
    }
  }
#else
  for (int k0 = 0; k0 < 128; ++k0) {
#pragma unroll
    for (int nt = 0; nt < 8; ++nt) {
#pragma unroll
      for (int r = 0; r < 8; ++r) {
        int row = wv * 16 + r + (lane < 16 ? 0 : 8);
        int col = nt * 16 + (lane & 15);
        acc[nt][r] += Ah[row * 128 + k0] * Bh[k0 * 128 + col];
      }
    }
  }
#endif

#pragma unroll
  for (int nt = 0; nt < 8; ++nt) {
#pragma unroll
    for (int r = 0; r < 8; ++r) {
      int row = wv * 16 + r + (lane < 16 ? 0 : 8);
      int col = nt * 16 + (lane & 15);
      float ab = acc[nt][r];
      float eye = (row == col) ? 1.f : 0.f;
      C1h[row * 128 + col] = s1 * ab + d1 * eye;
      if (C2h) C2h[row * 128 + col] = s2 * ab + d2 * eye;
    }
  }
}

// ================= LayerNorm (warp per row, 256 features) =================
__global__ void __launch_bounds__(256)
k_layernorm(const float* __restrict__ X, const float* __restrict__ g,
            const float* __restrict__ b, float* __restrict__ Y, int rows)
{
  int lane = threadIdx.x & 31, wv = threadIdx.x >> 5;
  int row = blockIdx.x * 8 + wv;
  if (row >= rows) return;
  const float* x = X + (long)row * DMODEL;
  float vals[8];
  float s = 0.f;
#pragma unroll
  for (int e = 0; e < 8; ++e) { vals[e] = x[lane + 32 * e]; s += vals[e]; }
  s = wsum(s);
  float mu = s * (1.f / 256.f);
  float q = 0.f;
#pragma unroll
  for (int e = 0; e < 8; ++e) { float d = vals[e] - mu; q += d * d; }
  q = wsum(q);
  float rstd = rsqrtf(q * (1.f / 256.f) + 1e-5f);
#pragma unroll
  for (int e = 0; e < 8; ++e) {
    int c = lane + 32 * e;
    Y[(long)row * DMODEL + c] = (vals[e] - mu) * rstd * g[c] + b[c];
  }
}

// ================= landmarks: mean over l=313 contiguous tokens =================
__global__ void __launch_bounds__(256)
k_landmarks(const float* __restrict__ qkv, float* __restrict__ ql, float* __restrict__ kl)
{
  int wg = blockIdx.x * 8 + (threadIdx.x >> 5);   // 512 warps total
  int lane = threadIdx.x & 31;
  int head = wg >> 7, m = wg & 127;
  float aq = 0.f, ak = 0.f;
  for (int j = 0; j < LCH; ++j) {
    long p = (long)(m * LCH + j) * 384;
    aq += qkv[p + head * 32 + lane];
    ak += qkv[p + 128 + head * 32 + lane];
  }
  ql[(head * 128 + m) * 32 + lane] = aq * (1.f / LCH);
  kl[(head * 128 + m) * 32 + lane] = ak * (1.f / LCH);
}

// ================= sim2 = q_l @ k_l^T (per head) =================
__global__ void __launch_bounds__(256)
k_sim2(const float* __restrict__ ql, const float* __restrict__ kl, float* __restrict__ a2)
{
  __shared__ __align__(16) float qs[128 * 32];
  __shared__ __align__(16) float ks[128 * 32];
  int head = blockIdx.x, tid = threadIdx.x;
  async_copy_f32(qs, ql + head * 4096, 4096, tid, 256);
  async_copy_f32(ks, kl + head * 4096, 4096, tid, 256);
  __syncthreads();
  for (int idx = tid; idx < 16384; idx += 256) {
    int i = idx >> 7, j = idx & 127;
    float s = 0.f;
#pragma unroll
    for (int d = 0; d < 32; ++d) s += qs[i * 32 + d] * ks[j * 32 + d];
    a2[head * 16384 + idx] = s;
  }
}

// ================= row softmax over 128 cols (warp per row) =================
__global__ void __launch_bounds__(256)
k_softmax128(float* __restrict__ Mx, int rows)
{
  int wv = threadIdx.x >> 5, lane = threadIdx.x & 31;
  int row = blockIdx.x * 8 + wv;
  if (row >= rows) return;
  float* p = Mx + (long)row * 128;
  float v[4]; float mx = -1e30f;
#pragma unroll
  for (int e = 0; e < 4; ++e) { v[e] = p[lane + 32 * e]; mx = fmaxf(mx, v[e]); }
  mx = wmaxr(mx);
  float s = 0.f;
#pragma unroll
  for (int e = 0; e < 4; ++e) { v[e] = __expf(v[e] - mx); s += v[e]; }
  s = wsum(s);
  float inv = 1.f / s;
#pragma unroll
  for (int e = 0; e < 4; ++e) p[lane + 32 * e] = v[e] * inv;
}

// ================= pinv init: z = a2^T / (max rowsum * max colsum) =================
__global__ void __launch_bounds__(256)
k_pinv_init(const float* __restrict__ a2, float* __restrict__ z)
{
  __shared__ float rs[128], cs[128];
  __shared__ float sinv;
  int head = blockIdx.x, tid = threadIdx.x;
  const float* a = a2 + head * 16384;
  if (tid < 128) {
    float s = 0.f;
    for (int j = 0; j < 128; ++j) s += fabsf(a[tid * 128 + j]);
    rs[tid] = s;
  } else {
    int c = tid - 128;
    float s = 0.f;
    for (int i = 0; i < 128; ++i) s += fabsf(a[i * 128 + c]);
    cs[c] = s;
  }
  __syncthreads();
  if (tid == 0) {
    float mr = 0.f, mc = 0.f;
    for (int i = 0; i < 128; ++i) { mr = fmaxf(mr, rs[i]); mc = fmaxf(mc, cs[i]); }
    sinv = 1.f / (mr * mc);
  }
  __syncthreads();
  float v = sinv;
  for (int idx = tid; idx < 16384; idx += 256) {
    int i = idx >> 7, j = idx & 127;
    z[head * 16384 + idx] = a[j * 128 + i] * v;
  }
}

// ======== sim3 row + streaming softmax + @v : a3v[head][i][32] ========
__global__ void __launch_bounds__(256)
k_sim3(const float* __restrict__ qkv, const float* __restrict__ ql, float* __restrict__ a3v)
{
  __shared__ float qv[32];
  __shared__ float red[8];
  __shared__ float vred[8][32];
  __shared__ float bmax, bsum;
  int head = blockIdx.y, i = blockIdx.x;
  int tid = threadIdx.x, lane = tid & 31, wv = tid >> 5;
  if (tid < 32) qv[tid] = ql[(head * 128 + i) * 32 + tid];
  __syncthreads();

  float mx = -1e30f;
  for (int j = tid; j < NPAD; j += 256) {
    const float* kr = qkv + (long)j * 384 + 128 + head * 32;
    float lg = 0.f;
#pragma unroll
    for (int d = 0; d < 32; ++d) lg += qv[d] * kr[d];
    mx = fmaxf(mx, lg);
  }
  mx = wmaxr(mx);
  if (lane == 0) red[wv] = mx;
  __syncthreads();
  if (tid == 0) {
    float m = -1e30f;
    for (int w = 0; w < 8; ++w) m = fmaxf(m, red[w]);
    bmax = m;
  }
  __syncthreads();
  float M0 = bmax;

  float s = 0.f;
  float vacc[32];
#pragma unroll
  for (int d = 0; d < 32; ++d) vacc[d] = 0.f;
  for (int j = tid; j < NPAD; j += 256) {
    const float* kr = qkv + (long)j * 384 + 128 + head * 32;
    const float* vr = qkv + (long)j * 384 + 256 + head * 32;
    float lg = 0.f;
#pragma unroll
    for (int d = 0; d < 32; ++d) lg += qv[d] * kr[d];
    float w = __expf(lg - M0);
    s += w;
#pragma unroll
    for (int d = 0; d < 32; ++d) vacc[d] += w * vr[d];
  }
  s = wsum(s);
  if (lane == 0) red[wv] = s;
#pragma unroll
  for (int d = 0; d < 32; ++d) {
    float t = wsum(vacc[d]);
    if (lane == 0) vred[wv][d] = t;
  }
  __syncthreads();
  if (tid == 0) {
    float t = 0.f;
    for (int w = 0; w < 8; ++w) t += red[w];
    bsum = t;
  }
  __syncthreads();
  if (tid < 32) {
    float t = 0.f;
    for (int w = 0; w < 8; ++w) t += vred[w][tid];
    a3v[(head * 128 + i) * 32 + tid] = t / bsum;
  }
}

// ================= w2 = z(pinv) @ a3v : [128,32] per head =================
__global__ void __launch_bounds__(256)
k_zw(const float* __restrict__ z, const float* __restrict__ a3v, float* __restrict__ w2)
{
  int head = blockIdx.x, tid = threadIdx.x;
  const float* zh = z + head * 16384;
  const float* ah = a3v + head * 4096;
  for (int o = tid; o < 4096; o += 256) {
    int i = o >> 5, d = o & 31;
    float s = 0.f;
    for (int m = 0; m < 128; ++m) s += zh[i * 128 + m] * ah[m * 32 + d];
    w2[head * 4096 + o] = s;
  }
}

// ======== sim1 softmax(128) @ w2 + depthwise seq conv(33) -> att inner ========
__global__ void __launch_bounds__(256)
k_sim1_out(const float* __restrict__ qkv, const float* __restrict__ kl,
           const float* __restrict__ w2, const float* __restrict__ resk,
           float* __restrict__ att)
{
  __shared__ __align__(16) float ks[4096];
  __shared__ __align__(16) float ws[4096];
  __shared__ float rk[33];
  int head = blockIdx.y;
  int tid = threadIdx.x, lane = tid & 31, wv = tid >> 5;
  async_copy_f32(ks, kl + head * 4096, 4096, tid, 256);
  async_copy_f32(ws, w2 + head * 4096, 4096, tid, 256);
  if (tid < 33) rk[tid] = resk[head * 33 + tid];
  __syncthreads();

  int p = blockIdx.x * 8 + wv;                 // token in [0, NPAD)
  float qd = qkv[(long)p * 384 + head * 32 + lane];
  float mx = -1e30f;
  for (int m = 0; m < 128; ++m) {
    float lg = wsum(qd * ks[m * 32 + lane]);
    mx = fmaxf(mx, lg);
  }
  float s = 0.f, acc = 0.f;
  for (int m = 0; m < 128; ++m) {
    float lg = wsum(qd * ks[m * 32 + lane]);
    float w = __expf(lg - mx);
    s += w;
    acc += w * ws[m * 32 + lane];
  }
  float out = acc / s;

  float res = 0.f;
  for (int a = 0; a < 33; ++a) {
    int j = p + a - 16;
    if (j >= 0 && j < NPAD)
      res += rk[a] * qkv[(long)j * 384 + 256 + head * 32 + lane];
  }
  att[(long)p * 128 + head * 32 + lane] = out + res;
}

// ================= CNN glue + fused depthwise 7/5/3 convs =================
__global__ void k_to_cnn(const float* __restrict__ h, float* __restrict__ cnn)
{
  int s = blockIdx.x, c = threadIdx.x;
  cnn[(long)c * 40000 + s] = h[(long)(1 + s) * DMODEL + c];
}
__global__ void k_from_cnn(const float* __restrict__ cnn, float* __restrict__ h)
{
  int s = blockIdx.x, c = threadIdx.x;
  h[(long)(1 + s) * DMODEL + c] = cnn[(long)c * 40000 + s];
}
__global__ void k_dwconv(const float* __restrict__ cin,
                         const float* __restrict__ w7, const float* __restrict__ b7,
                         const float* __restrict__ w5, const float* __restrict__ b5,
                         const float* __restrict__ w3, const float* __restrict__ b3,
                         float* __restrict__ cout)
{
  int s = blockIdx.x, c = threadIdx.x;
  int y = s / SIDE, x = s % SIDE;
  const float* pc = cin + (long)c * 40000;
  float acc = pc[s];

  float a7 = 0.f;
  for (int a = 0; a < 7; ++a) {
    int yy = y + a - 3; if (yy < 0 || yy >= SIDE) continue;
    for (int bb = 0; bb < 7; ++bb) {
      int xx = x + bb - 3; if (xx < 0 || xx >= SIDE) continue;
      a7 += pc[yy * SIDE + xx] * w7[c * 49 + a * 7 + bb];
    }
  }
  acc += a7 + b7[c];

  float a5 = 0.f;
  for (int a = 0; a < 5; ++a) {
    int yy = y + a - 2; if (yy < 0 || yy >= SIDE) continue;
    for (int bb = 0; bb < 5; ++bb) {
      int xx = x + bb - 2; if (xx < 0 || xx >= SIDE) continue;
      a5 += pc[yy * SIDE + xx] * w5[c * 25 + a * 5 + bb];
    }
  }
  acc += a5 + b5[c];

  float a3 = 0.f;
  for (int a = 0; a < 3; ++a) {
    int yy = y + a - 1; if (yy < 0 || yy >= SIDE) continue;
    for (int bb = 0; bb < 3; ++bb) {
      int xx = x + bb - 1; if (xx < 0 || xx >= SIDE) continue;
      a3 += pc[yy * SIDE + xx] * w3[c * 9 + a * 3 + bb];
    }
  }
  acc += a3 + b3[c];

  cout[(long)c * 40000 + s] = acc;
}

// ================= cls write + final LN + fc2 =================
__global__ void k_addcls(const float* __restrict__ cls, float* __restrict__ h)
{
  h[threadIdx.x] = cls[threadIdx.x];
}

__global__ void __launch_bounds__(256)
k_final(const float* __restrict__ h, const float* __restrict__ g, const float* __restrict__ b,
        const float* __restrict__ w, const float* __restrict__ bias, float* __restrict__ out)
{
  __shared__ float sh[256];
  __shared__ float ln[256];
  int tid = threadIdx.x;
  float x = h[tid];
  sh[tid] = x; __syncthreads();
  for (int st = 128; st > 0; st >>= 1) { if (tid < st) sh[tid] += sh[tid + st]; __syncthreads(); }
  float mu = sh[0] * (1.f / 256.f);
  __syncthreads();
  float d = x - mu;
  sh[tid] = d * d; __syncthreads();
  for (int st = 128; st > 0; st >>= 1) { if (tid < st) sh[tid] += sh[tid + st]; __syncthreads(); }
  float rstd = rsqrtf(sh[0] * (1.f / 256.f) + 1e-5f);
  __syncthreads();
  ln[tid] = d * rstd * g[tid] + b[tid];
  __syncthreads();
  if (tid < 4) {
    float s = bias[tid];
    for (int dd = 0; dd < 256; ++dd) s += ln[dd] * w[dd * 4 + tid];
    out[tid] = s;
  }
}

// ================= host-side sequencing =================
static void attention_block(const float* lng, const float* lnb, const float* qkvw,
                            const float* outw, const float* outb, const float* resk,
                            float* h, float* lnbuf, float* qkvbuf, float* attbuf,
                            float* ql, float* kl, float* a2, float* za, float* zb,
                            float* xz, float* u1, float* u2, float* u3,
                            float* a3v, float* w2, hipStream_t stream)
{
  k_layernorm<<<(NS + 7) / 8, 256, 0, stream>>>(h, lng, lnb, lnbuf, NS);
  // qkv = LN(h) @ W, padded to NPAD rows (leading 63 zero rows), q pre-scaled
  k_gemm_f16<<<dim3(NPAD / 128, 3), 256, 0, stream>>>(
      lnbuf, NS, PADF, DMODEL, qkvw, 384, nullptr,
      qkvbuf, 384, nullptr, 0, 0, NPAD, 384, DMODEL, 0, QSCALE, 128);
  k_landmarks<<<64, 256, 0, stream>>>(qkvbuf, ql, kl);
  k_sim2<<<4, 256, 0, stream>>>(ql, kl, a2);
  k_softmax128<<<64, 256, 0, stream>>>(a2, 512);         // 4 heads * 128 rows
  k_pinv_init<<<4, 256, 0, stream>>>(a2, za);
  float* zc = za; float* zn = zb;
  for (int it = 0; it < 6; ++it) {
    // z = 0.25 * z @ (13I - xz @ (15I - xz @ (7I - xz))),  xz = a2 @ z
    k_mm128<<<4, 256, 0, stream>>>(a2, zc, xz, 1.f, 0.f, u1, -1.f, 7.f);
    k_mm128<<<4, 256, 0, stream>>>(xz, u1, u2, -1.f, 15.f, nullptr, 0.f, 0.f);
    k_mm128<<<4, 256, 0, stream>>>(xz, u2, u3, -1.f, 13.f, nullptr, 0.f, 0.f);
    k_mm128<<<4, 256, 0, stream>>>(zc, u3, zn, 0.25f, 0.f, nullptr, 0.f, 0.f);
    float* t = zc; zc = zn; zn = t;
  }
  k_sim3<<<dim3(128, 4), 256, 0, stream>>>(qkvbuf, ql, a3v);
  k_zw<<<4, 256, 0, stream>>>(zc, a3v, w2);
  k_sim1_out<<<dim3(NPAD / 8, 4), 256, 0, stream>>>(qkvbuf, kl, w2, resk, attbuf);
  // h[p-PADF] += att @ out_w + out_b   (drops the 63 pad rows)
  k_gemm_f16<<<dim3(NPAD / 128, 2), 256, 0, stream>>>(
      attbuf, NPAD, 0, 128, outw, DMODEL, outb,
      nullptr, 0, h, PADF, DMODEL, NPAD, DMODEL, 128, 0, 1.f, 0);
}

extern "C" void kernel_launch(void* const* d_in, const int* in_sizes, int n_in,
                              void* d_out, int out_size, void* d_ws, size_t ws_size,
                              hipStream_t stream)
{
  (void)in_sizes; (void)n_in; (void)out_size; (void)ws_size;
  const float* x_path = (const float*)d_in[0];
  const float* fc1_w  = (const float*)d_in[1];
  const float* fc1_b  = (const float*)d_in[2];
  const float* cls    = (const float*)d_in[3];
  const float* ln1_g  = (const float*)d_in[4];
  const float* ln1_b  = (const float*)d_in[5];
  const float* qkv1_w = (const float*)d_in[6];
  const float* out1_w = (const float*)d_in[7];
  const float* out1_b = (const float*)d_in[8];
  const float* res1_k = (const float*)d_in[9];
  const float* p7_w   = (const float*)d_in[10];
  const float* p7_b   = (const float*)d_in[11];
  const float* p5_w   = (const float*)d_in[12];
  const float* p5_b   = (const float*)d_in[13];
  const float* p3_w   = (const float*)d_in[14];
  const float* p3_b   = (const float*)d_in[15];
  const float* ln2_g  = (const float*)d_in[16];
  const float* ln2_b  = (const float*)d_in[17];
  const float* qkv2_w = (const float*)d_in[18];
  const float* out2_w = (const float*)d_in[19];
  const float* out2_b = (const float*)d_in[20];
  const float* res2_k = (const float*)d_in[21];
  const float* lnf_g  = (const float*)d_in[22];
  const float* lnf_b  = (const float*)d_in[23];
  const float* fc2_w  = (const float*)d_in[24];
  const float* fc2_b  = (const float*)d_in[25];

  float* ws = (float*)d_ws;
  float* h      = ws;                              // NS*256
  float* lnbuf  = h + (long)NS * DMODEL;           // NS*256  (CNN input aliases this)
  float* qkvbuf = lnbuf + (long)NS * DMODEL;       // NPAD*384 (CNN output aliases this)
  float* attbuf = qkvbuf + (long)NPAD * 384;       // NPAD*128
  float* small  = attbuf + (long)NPAD * 128;
  float* ql  = small;          // 4*128*32
  float* kl  = ql  + 16384;
  float* a2  = kl  + 16384;    // 4*128*128
  float* za  = a2  + 65536;
  float* zb  = za  + 65536;
  float* xz  = zb  + 65536;
  float* u1  = xz  + 65536;
  float* u2  = u1  + 65536;
  float* u3  = u2  + 65536;
  float* a3v = u3  + 65536;    // 4*128*32
  float* w2  = a3v + 16384;    // 4*128*32

  // stage 1: h[1..40000] = relu(x_path @ fc1_w + fc1_b); h[0] = cls
  k_gemm_f16<<<dim3((NTOK + 127) / 128, 2), 256, 0, stream>>>(
      x_path, NTOK, 0, 1024, fc1_w, DMODEL, fc1_b,
      h + DMODEL, DMODEL, nullptr, 0, 0, NTOK, DMODEL, 1024, 1, 1.f, 0);
  k_addcls<<<1, 256, 0, stream>>>(cls, h);

  // block 1
  attention_block(ln1_g, ln1_b, qkv1_w, out1_w, out1_b, res1_k,
                  h, lnbuf, qkvbuf, attbuf, ql, kl, a2, za, zb, xz, u1, u2, u3,
                  a3v, w2, stream);

  // CNN: feat -> NCHW, fused identity + dw7 + dw5 + dw3, back to tokens
  k_to_cnn<<<40000, 256, 0, stream>>>(h, lnbuf);
  k_dwconv<<<40000, 256, 0, stream>>>(lnbuf, p7_w, p7_b, p5_w, p5_b, p3_w, p3_b, qkvbuf);
  k_from_cnn<<<40000, 256, 0, stream>>>(qkvbuf, h);

  // block 2
  attention_block(ln2_g, ln2_b, qkv2_w, out2_w, out2_b, res2_k,
                  h, lnbuf, qkvbuf, attbuf, ql, kl, a2, za, zb, xz, u1, u2, u3,
                  a3v, w2, stream);

  // final LN on cls row + fc2
  k_final<<<1, 256, 0, stream>>>(h, lnf_g, lnf_b, fc2_w, fc2_b, (float*)d_out);
}
